// GPT2Attention_classic_70300024701796
// MI455X (gfx1250) — matrix operand order repaired and checked by reading, not compiled, not run on hardware
//
#include <hip/hip_runtime.h>
#include <hip/hip_bf16.h>

typedef __attribute__((ext_vector_type(16))) __bf16 bf16x16;
typedef __attribute__((ext_vector_type(8)))  float  f32x8;

constexpr int NB = 2, NS = 2048, NE = 1024, NH = 16, NHD = 64;

#define DEV __device__ __forceinline__

// Tensor Data Mover path (this toolchain exposes the 6-arg clang-23 builtin:
// (u32x4 g0, i32x8 g1, i32x4 g2, i32x4 g3, i32x8 g4, i32 cpol)).
#if defined(__gfx1250__) && __has_builtin(__builtin_amdgcn_tensor_load_to_lds) && \
    __has_builtin(__builtin_amdgcn_s_wait_tensorcnt)
#define USE_TDM 1
#else
#define USE_TDM 0
#endif

union FragU {
  bf16x16 v;
  __bf16  h[16];
};

union Pack8 {
  uint4  u;
  __bf16 h[8];
};

// A-fragment loader (16xK tile; this lane holds row m = lane&15, kh = lane>>4).
// Per ISA 16-bit A layout: h[0..7] = K{k0+kh*8+0..7}, h[8..15] = K{k0+16+kh*8+0..7}.
template <typename T>
DEV bf16x16 load_a(const T* row, int kh, int k0) {
  FragU f;
  const T* p0 = row + k0 + kh * 8;
  const T* p1 = row + k0 + 16 + kh * 8;
#pragma unroll
  for (int i = 0; i < 8; ++i) f.h[i] = (__bf16)(float)p0[i];
#pragma unroll
  for (int i = 0; i < 8; ++i) f.h[8 + i] = (__bf16)(float)p1[i];
  return f.v;
}

// B-fragment loader (Kx16 tile; this lane holds column n = lane&15, kh = lane>>4).
// Per ISA 16-bit B layout: h[0..7] = K{k0+kh*8+0..7}, h[8..15] = K{k0+16+kh*8+0..7}.
template <typename T>
DEV bf16x16 load_b(const T* col, int kh, int k0) {
  FragU f;
  const T* p0 = col + k0 + kh * 8;
  const T* p1 = col + k0 + 16 + kh * 8;
#pragma unroll
  for (int i = 0; i < 8; ++i) f.h[i] = (__bf16)(float)p0[i];
#pragma unroll
  for (int i = 0; i < 8; ++i) f.h[8 + i] = (__bf16)(float)p1[i];
  return f.v;
}

DEV f32x8 wmma_bf16(bf16x16 a, bf16x16 b, f32x8 c) {
  // (neg_a, A, neg_b, B, c_mod, C, reuse_a, reuse_b)
  return __builtin_amdgcn_wmma_f32_16x16x32_bf16(false, a, false, b, (short)0, c,
                                                 false, false);
}

#if USE_TDM
typedef __attribute__((ext_vector_type(4))) unsigned int u32x4;
typedef __attribute__((ext_vector_type(8))) int i32x8;
typedef __attribute__((ext_vector_type(4))) int i32x4;

// 2-D TDM load: `rows` rows of `rowElems` bf16 (row pitch `strideElems`) into LDS.
// D# built per CDNA5 ISA 8.3/8.4 (Group0: count/lds_addr/global_addr/type=2;
// Group1: data_size=2B, tensor dims, tile dims, dim0 stride).
DEV void tdm_load_2d(unsigned lds_off, const __bf16* gptr, unsigned rowElems,
                     unsigned rows, unsigned strideElems) {
  const unsigned long long ga = (unsigned long long)(size_t)gptr;
  u32x4 g0;
  g0[0] = 1u;                                          // count=1
  g0[1] = lds_off;                                     // lds_addr (bytes)
  g0[2] = (unsigned)(ga & 0xffffffffu);                // global_addr[31:0]
  g0[3] = (unsigned)((ga >> 32) & 0x1ffffffu) | (2u << 30);  // addr[56:32]|type=2
  i32x8 g1;
  g1[0] = (1 << 16);                    // workgroup_mask=0, data_size=1 (2 bytes)
  g1[1] = (int)(rowElems << 16);        // tensor_dim0[15:0] in bits [31:16]
  g1[2] = (int)(rows << 16);            // tensor_dim0 hi=0 | tensor_dim1[15:0]
  g1[3] = (int)(rowElems << 16);        // tensor_dim1 hi=0 | tile_dim0
  g1[4] = (int)rows;                    // tile_dim1 | tile_dim2=0
  g1[5] = (int)strideElems;             // tensor_dim0_stride[31:0]
  g1[6] = 0;                            // stride hi | tensor_dim1_stride lo
  g1[7] = 0;
  const i32x4 z4 = {0, 0, 0, 0};
  const i32x8 z8 = {0, 0, 0, 0, 0, 0, 0, 0};
  __builtin_amdgcn_tensor_load_to_lds(g0, g1, z4, z4, z8, 0);
}
#endif

// ---------------------------------------------------------------------------
// Kernel 1: per-head QKV projection. One wave per (16-row tile, head).
// Q,K stored [b,h,s,d] bf16; V stored transposed [b,h,d,s] bf16 (b128 stores).
// ---------------------------------------------------------------------------
template <bool TRANSPOSED>
DEV void head_gemm_store(bf16x16 a0, bf16x16 a1, const float* __restrict__ W,
                         const float* __restrict__ bias, __bf16* __restrict__ dst,
                         int s0, int n, int kh) {
#pragma unroll
  for (int nt = 0; nt < 4; ++nt) {
    const float* wrow = W + (size_t)(nt * 16 + n) * NHD;  // W[n_out][k]
    f32x8 acc = {};
    acc = wmma_bf16(a0, load_b(wrow, kh, 0), acc);
    acc = wmma_bf16(a1, load_b(wrow, kh, 32), acc);
    const float bv = bias[nt * 16 + n];
    if constexpr (TRANSPOSED) {
      // dst layout [d][s]: this lane's 8 rows are contiguous in s -> one b128.
      Pack8 p;
#pragma unroll
      for (int r = 0; r < 8; ++r) p.h[r] = (__bf16)(acc[r] + bv);
      *reinterpret_cast<uint4*>(dst + (size_t)(nt * 16 + n) * NS + s0 + 8 * kh) =
          p.u;
    } else {
#pragma unroll
      for (int r = 0; r < 8; ++r) {
        const int srow = s0 + r + 8 * kh;  // C/D layout: m = r + 8*(lane>=16)
        dst[(size_t)srow * NHD + nt * 16 + n] = (__bf16)(acc[r] + bv);
      }
    }
  }
}

__global__ __launch_bounds__(256) void qkv_kernel(
    const float* __restrict__ x,
    const float* __restrict__ Wq, const float* __restrict__ bq,
    const float* __restrict__ Wk, const float* __restrict__ bk,
    const float* __restrict__ Wv, const float* __restrict__ bv,
    __bf16* __restrict__ Q, __bf16* __restrict__ K, __bf16* __restrict__ Vt) {
  const int lane = threadIdx.x & 31;
  const int wave = blockIdx.x * (blockDim.x >> 5) + (threadIdx.x >> 5);
  const int h    = wave & (NH - 1);
  const int tile = wave >> 4;            // 0 .. NB*NS/16-1
  const int b    = tile >> 7;            // tile / (NS/16)
  const int s0   = (tile & 127) * 16;
  const int n    = lane & 15;
  const int kh   = lane >> 4;

  const float* xrow = x + ((size_t)(b * NS + s0 + n) * NE + h * NHD);
  const bf16x16 a0 = load_a(xrow, kh, 0);
  const bf16x16 a1 = load_a(xrow, kh, 32);

  const size_t bh = (size_t)(b * NH + h);
  head_gemm_store<false>(a0, a1, Wq, bq, Q + bh * NS * NHD, s0, n, kh);
  head_gemm_store<false>(a0, a1, Wk, bk, K + bh * NS * NHD, s0, n, kh);
  head_gemm_store<true >(a0, a1, Wv, bv, Vt + bh * NHD * NS, s0, n, kh);
}

// ---------------------------------------------------------------------------
// Kernel 2: causal flash attention. One wave per (b, h, 32-row q tile).
// KV tiles of 32; K/V tiles staged into LDS by the Tensor Data Mover and
// shared by both 16-row A-fragment groups. P tile (16x32) is exactly one
// A-fragment, rearranged D-layout -> A-layout through LDS.
// ---------------------------------------------------------------------------
__global__ __launch_bounds__(32) void fa_kernel(
    const __bf16* __restrict__ Q, const __bf16* __restrict__ K,
    const __bf16* __restrict__ Vt, __bf16* __restrict__ AO) {
#if USE_TDM
  __shared__ __bf16 ldsK[32 * NHD];   // K tile:  32 rows x 64 (4 KB)
  __shared__ __bf16 ldsV[NHD * 32];   // V^T tile: 64 rows x 32 (4 KB)
#endif
  __shared__ __bf16 ldsP[2 * 16 * 32];  // P tiles for both q sub-tiles (2 KB)

  const int lane = threadIdx.x;
  const int n  = lane & 15;
  const int kh = lane >> 4;
  const int wid = blockIdx.x;    // (b*NH + h) * (NS/32) + qt
  const int qt = wid & 63;       // 32-row q tile index (NS/32 = 64)
  const int bh = wid >> 6;
  const int h  = bh & (NH - 1);
  const int b  = bh >> 4;

  const __bf16* Qb = Q  + (size_t)bh * NS * NHD;
  const __bf16* Kb = K  + (size_t)bh * NS * NHD;
  const __bf16* Vb = Vt + (size_t)bh * NHD * NS;

  bf16x16 aq[2][2];
#pragma unroll
  for (int at = 0; at < 2; ++at) {
    const __bf16* qrow = Qb + (size_t)(qt * 32 + at * 16 + n) * NHD;
    aq[at][0] = load_a(qrow, kh, 0);
    aq[at][1] = load_a(qrow, kh, 32);
  }

  f32x8 o[2][4] = {};
  float rmax[2][8], rsum[2][8];
#pragma unroll
  for (int at = 0; at < 2; ++at)
#pragma unroll
    for (int r = 0; r < 8; ++r) { rmax[at][r] = -3.0e38f; rsum[at][r] = 0.0f; }

  for (int j = 0; j <= qt; ++j) {
    const int kv = j * 32;

#if USE_TDM
    // Ensure last iteration's LDS fragment reads retired before TDM overwrite.
    asm volatile("s_wait_dscnt 0x0" ::: "memory");
    tdm_load_2d((unsigned)(size_t)&ldsK[0], Kb + (size_t)kv * NHD, NHD, 32, NHD);
    tdm_load_2d((unsigned)(size_t)&ldsV[0], Vb + kv, 32, NHD, NS);
    __builtin_amdgcn_s_wait_tensorcnt(0);
    asm volatile("" ::: "memory");  // TDM wrote LDS behind the compiler's back
#endif

#pragma unroll
    for (int at = 0; at < 2; ++at) {
      // S = Q @ K^T for a 16x32 score tile (two 16x16 accumulators).
      f32x8 c0 = {}, c1 = {};
#if USE_TDM
      const __bf16* kr0 = ldsK + (size_t)n * NHD;
      const __bf16* kr1 = ldsK + (size_t)(16 + n) * NHD;
#else
      const __bf16* kr0 = Kb + (size_t)(kv + n) * NHD;
      const __bf16* kr1 = Kb + (size_t)(kv + 16 + n) * NHD;
#endif
      c0 = wmma_bf16(aq[at][0], load_b(kr0, kh, 0),  c0);
      c0 = wmma_bf16(aq[at][1], load_b(kr0, kh, 32), c0);
      c1 = wmma_bf16(aq[at][0], load_b(kr1, kh, 0),  c1);
      c1 = wmma_bf16(aq[at][1], load_b(kr1, kh, 32), c1);

      // Scale, causal mask (only the diagonal tile needs it), online softmax.
#pragma unroll
      for (int r = 0; r < 8; ++r) {
        const int qi = qt * 32 + at * 16 + r + 8 * kh;
        float s0v = c0[r] * 0.125f;  // 1/sqrt(64)
        float s1v = c1[r] * 0.125f;
        if (kv + n      > qi) s0v = -3.0e38f;
        if (kv + 16 + n > qi) s1v = -3.0e38f;

        float vmax = fmaxf(s0v, s1v);
#pragma unroll
        for (int off = 1; off < 16; off <<= 1)
          vmax = fmaxf(vmax, __shfl_xor(vmax, off, 32));
        const float mn = fmaxf(rmax[at][r], vmax);

        const float p0 = __expf(s0v - mn);
        const float p1 = __expf(s1v - mn);
        float ps = p0 + p1;
#pragma unroll
        for (int off = 1; off < 16; off <<= 1) ps += __shfl_xor(ps, off, 32);

        const float corr = __expf(rmax[at][r] - mn);
        rsum[at][r] = rsum[at][r] * corr + ps;
        rmax[at][r] = mn;
        o[at][0][r] *= corr; o[at][1][r] *= corr;
        o[at][2][r] *= corr; o[at][3][r] *= corr;

        const int m = r + 8 * kh;
        ldsP[at * 512 + m * 32 + n]      = (__bf16)p0;
        ldsP[at * 512 + m * 32 + 16 + n] = (__bf16)p1;
      }
    }
    __syncthreads();  // single-wave WG: orders LDS P writes -> A-frag reads
    const bf16x16 ap0 = load_a(ldsP + (size_t)n * 32, kh, 0);
    const bf16x16 ap1 = load_a(ldsP + 512 + (size_t)n * 32, kh, 0);
    __syncthreads();  // reads done before next iteration overwrites ldsP

    // O += P @ V; each V B-fragment is shared by both q sub-tiles.
#pragma unroll
    for (int dt = 0; dt < 4; ++dt) {
#if USE_TDM
      const __bf16* vrow = ldsV + (size_t)(dt * 16 + n) * 32;
#else
      const __bf16* vrow = Vb + (size_t)(dt * 16 + n) * NS + kv;
#endif
      const bf16x16 bv = load_b(vrow, kh, 0);
      o[0][dt] = wmma_bf16(ap0, bv, o[0][dt]);
      o[1][dt] = wmma_bf16(ap1, bv, o[1][dt]);
    }
  }

  // Normalize and store to attention output [B*S, E], e = h*64 + dt*16 + n.
#pragma unroll
  for (int at = 0; at < 2; ++at)
#pragma unroll
    for (int r = 0; r < 8; ++r) {
      const float inv = 1.0f / rsum[at][r];
      const int srow = qt * 32 + at * 16 + r + 8 * kh;
      __bf16* orow = AO + ((size_t)(b * NS + srow) * NE + h * NHD);
#pragma unroll
      for (int dt = 0; dt < 4; ++dt)
        orow[dt * 16 + n] = (__bf16)(o[at][dt][r] * inv);
    }
}

// ---------------------------------------------------------------------------
// Kernel 3: output projection out = AO @ Wp^T + bp  (4096 x 1024 x 1024 GEMM).
// One wave per 16x64 output tile (A-frag reused across 4 N-tiles).
// ---------------------------------------------------------------------------
__global__ __launch_bounds__(256) void proj_kernel(
    const __bf16* __restrict__ AO, const float* __restrict__ W,
    const float* __restrict__ bias, float* __restrict__ out) {
  const int lane = threadIdx.x & 31;
  const int wave = blockIdx.x * (blockDim.x >> 5) + (threadIdx.x >> 5);
  const int ng = wave & 15;   // column group of 64 (NE/64 = 16)
  const int mt = wave >> 4;   // 16-row tile (NB*NS/16 = 256)
  const int n  = lane & 15;
  const int kh = lane >> 4;

  const __bf16* arow = AO + (size_t)(mt * 16 + n) * NE;
  f32x8 acc[4] = {};
  for (int k0 = 0; k0 < NE; k0 += 32) {
    __builtin_prefetch(arow + k0 + 128, 0, 1);  // global_prefetch_b8
    const bf16x16 a = load_a(arow, kh, k0);
#pragma unroll
    for (int nt = 0; nt < 4; ++nt) {
      const float* wrow = W + (size_t)(ng * 64 + nt * 16 + n) * NE;  // Wp[n][k]
      acc[nt] = wmma_bf16(a, load_b(wrow, kh, k0), acc[nt]);
    }
  }
#pragma unroll
  for (int nt = 0; nt < 4; ++nt) {
    const int col = ng * 64 + nt * 16 + n;
    const float bv = bias[col];
#pragma unroll
    for (int r = 0; r < 8; ++r)
      out[(size_t)(mt * 16 + r + 8 * kh) * NE + col] = acc[nt][r] + bv;
  }
}

// ---------------------------------------------------------------------------
extern "C" void kernel_launch(void* const* d_in, const int* in_sizes, int n_in,
                              void* d_out, int out_size, void* d_ws, size_t ws_size,
                              hipStream_t stream) {
  const float* x  = (const float*)d_in[0];
  const float* Wq = (const float*)d_in[1];
  const float* bq = (const float*)d_in[2];
  const float* Wk = (const float*)d_in[3];
  const float* bk = (const float*)d_in[4];
  const float* Wv = (const float*)d_in[5];
  const float* bv = (const float*)d_in[6];
  const float* Wp = (const float*)d_in[7];
  const float* bp = (const float*)d_in[8];
  float* out = (float*)d_out;

  constexpr size_t QKV_ELEMS = (size_t)NB * NH * NS * NHD;  // 4M bf16 each
  __bf16* Q  = (__bf16*)d_ws;
  __bf16* Kt = Q  + QKV_ELEMS;
  __bf16* Vt = Kt + QKV_ELEMS;
  __bf16* AO = Vt + QKV_ELEMS;   // [B*S, E] bf16; total ws use = 32 MB

  (void)in_sizes; (void)n_in; (void)out_size; (void)ws_size;

  // 1) QKV: (NB*NS/16)*NH = 4096 waves, 8 waves/block.
  qkv_kernel<<<(NB * NS / 16) * NH / 8, 256, 0, stream>>>(
      x, Wq, bq, Wk, bk, Wv, bv, Q, Kt, Vt);

  // 2) Flash attention: one 32-thread block (one wave) per (b,h, 32-row qtile).
  fa_kernel<<<NB * NH * (NS / 32), 32, 0, stream>>>(Q, Kt, Vt, AO);

  // 3) Output projection: 256 row tiles x 16 col groups = 4096 waves.
  proj_kernel<<<(NB * NS / 16) * (NE / 64) / 8, 256, 0, stream>>>(AO, Wp, bp, out);
}
